// Projection_12506944766612
// MI455X (gfx1250) — matrix operand order repaired and checked by reading, not compiled, not run on hardware
//
#include <hip/hip_runtime.h>
#include <hip/hip_bf16.h>

typedef __attribute__((ext_vector_type(8)))  float   v8f;
typedef __attribute__((ext_vector_type(8)))  __bf16  v8bf;
typedef __attribute__((ext_vector_type(16))) __bf16  v16bf;

// ---------------------------------------------------------------------------
// WMMA helpers (CDNA5 gfx1250, wave32, D = A(16x32 bf16) x B(32x16 bf16) + C)
// A layout (this code's k order, not the ISA table order): lane r=lane&15 -> row M=r; hi=lane>>4;
//   per-lane 16 contiguous vals: K = k0 + hi*16 + {0..15}
// B layout (same k order as A): lane r -> col N=r; per-lane 16 contiguous
//   vals: K = k0 + hi*16 + {0..15}
// C/D layout: lane r -> N=r; VGPR i -> M = i + hi*8
// ---------------------------------------------------------------------------
__device__ __forceinline__ v8f wmma_bf16(v16bf a, v16bf b, v8f c) {
  return __builtin_amdgcn_wmma_f32_16x16x32_bf16(false, a, false, b, (short)0, c,
                                                 false, false);
}

__device__ __forceinline__ v16bf load_a_frag(const __bf16* p) {
  // p already offset by hi*16; 16 contiguous elements
  v8bf lo = *(const v8bf*)(p);
  v8bf hi = *(const v8bf*)(p + 8);
  return __builtin_shufflevector(lo, hi, 0, 1, 2, 3, 4, 5, 6, 7, 8, 9, 10, 11,
                                 12, 13, 14, 15);
}

__device__ __forceinline__ v16bf load_b_frag(const __bf16* p) {
  // p already offset by hi*16; 16 contiguous elements
  v8bf lo = *(const v8bf*)(p);
  v8bf hi = *(const v8bf*)(p + 8);
  return __builtin_shufflevector(lo, hi, 0, 1, 2, 3, 4, 5, 6, 7, 8, 9, 10, 11,
                                 12, 13, 14, 15);
}

__device__ __forceinline__ void store8(float* p, v8f c) {
  *(float4*)(p)     = make_float4(c[0], c[1], c[2], c[3]);
  *(float4*)(p + 4) = make_float4(c[4], c[5], c[6], c[7]);
}

// ---------------------------------------------------------------------------
// f32 -> bf16 elementwise convert (weights)
// ---------------------------------------------------------------------------
__global__ void __launch_bounds__(256) cvt_bf16_kernel(
    const float* __restrict__ in, __bf16* __restrict__ out, int n) {
  int i = blockIdx.x * 256 + threadIdx.x;
  if (i < n) out[i] = (__bf16)in[i];
}

// ---------------------------------------------------------------------------
// (B,C,L) f32 -> (B,L,C) bf16 (+ optional f32 copy), tiled LDS transpose
// ---------------------------------------------------------------------------
__global__ void __launch_bounds__(256) transpose_cvt_kernel(
    const float* __restrict__ in, float* __restrict__ outf,
    __bf16* __restrict__ outb, int C, int L) {
  __shared__ float t[32][33];
  int b = blockIdx.z;
  int l0 = blockIdx.x * 32, c0 = blockIdx.y * 32;
  const float* inp = in + (size_t)b * C * L;
#pragma unroll
  for (int k = 0; k < 4; ++k) {
    int c = c0 + threadIdx.y + k * 8;
    t[threadIdx.y + k * 8][threadIdx.x] = inp[(size_t)c * L + l0 + threadIdx.x];
  }
  __syncthreads();
#pragma unroll
  for (int k = 0; k < 4; ++k) {
    int l = l0 + threadIdx.y + k * 8;
    float v = t[threadIdx.x][threadIdx.y + k * 8];
    size_t idx = ((size_t)b * L + l) * C + c0 + threadIdx.x;
    if (outf) outf[idx] = v;
    outb[idx] = (__bf16)v;
  }
}

// ---------------------------------------------------------------------------
// GEMM: Out[b][n][m] = sum_k W[m][k] * Xt[b][n][k]
// W: (M,K) bf16 row-major.  Xt: (B,N,K) bf16 row-major.  Out: (B,N,M) f32.
// 8 waves / block; each wave computes a 16(M) x 64(N) tile.
// (Bias omitted: every conv output feeds InstanceNorm -> bias is a no-op.)
// ---------------------------------------------------------------------------
__global__ void __launch_bounds__(256) gemm_bf16_kernel(
    const __bf16* __restrict__ W, const __bf16* __restrict__ Xt,
    float* __restrict__ Out, int M, int N, int K) {
  int tid = threadIdx.x;
  int wave = tid >> 5, lane = tid & 31, r = lane & 15, hi = lane >> 4;
  int mt = M >> 4, ng = N >> 6;
  int per_b = mt * ng;
  int task = blockIdx.x * 8 + wave;
  int b = task / per_b;
  int rem = task % per_b;
  int mbase = (rem / ng) * 16;
  int nbase = (rem % ng) * 64;

  const __bf16* arow = W + (size_t)(mbase + r) * K + hi * 16;
  const __bf16* brow = Xt + ((size_t)b * N + nbase + r) * K + hi * 16;

  v8f acc0 = {}, acc1 = {}, acc2 = {}, acc3 = {};
  for (int k0 = 0; k0 < K; k0 += 32) {
    v16bf a = load_a_frag(arow + k0);
    acc0 = wmma_bf16(a, load_b_frag(brow + k0), acc0);
    acc1 = wmma_bf16(a, load_b_frag(brow + (size_t)16 * K + k0), acc1);
    acc2 = wmma_bf16(a, load_b_frag(brow + (size_t)32 * K + k0), acc2);
    acc3 = wmma_bf16(a, load_b_frag(brow + (size_t)48 * K + k0), acc3);
  }
  float* orow = Out + ((size_t)b * N + nbase + r) * M + mbase + hi * 8;
  store8(orow, acc0);
  store8(orow + (size_t)16 * M, acc1);
  store8(orow + (size_t)32 * M, acc2);
  store8(orow + (size_t)48 * M, acc3);
}

// ---------------------------------------------------------------------------
// InstanceNorm over L per (b, channel).  X: (B, L, C) f32.
// MODE 0: outb (B,L,C) bf16                       (q, k)
// MODE 1: outb (B,C,L) bf16                       (v, channel-major for PV gemm)
// MODE 2: outb (B,L,C) bf16 = norm(X) + addsrc    (proj + x_query residual)
// MODE 3: outb (B,L,C) bf16 = relu(norm(X))       (ff hidden)
// MODE 4: outf (B,C,L) f32                        (final output)
// ---------------------------------------------------------------------------
template <int MODE>
__global__ void __launch_bounds__(256) inorm_kernel(
    const float* __restrict__ X, const float* __restrict__ addsrc,
    float* __restrict__ outf, __bf16* __restrict__ outb, int C) {
  constexpr int L = 1024;
  int c = blockIdx.x * 256 + threadIdx.x;
  int b = blockIdx.y;
  const float* col = X + (size_t)b * L * C + c;
  float s1 = 0.f, s2 = 0.f;
  for (int l = 0; l < L; ++l) {
    float v = col[(size_t)l * C];
    s1 += v;
    s2 += v * v;
  }
  float mean = s1 * (1.0f / L);
  float var = s2 * (1.0f / L) - mean * mean;
  float rs = rsqrtf(var + 1e-5f);
  for (int l = 0; l < L; ++l) {
    float v = (col[(size_t)l * C] - mean) * rs;
    size_t lc = ((size_t)b * L + l) * C + c;
    if constexpr (MODE == 0) {
      outb[lc] = (__bf16)v;
    } else if constexpr (MODE == 1) {
      outb[((size_t)b * C + c) * L + l] = (__bf16)v;
    } else if constexpr (MODE == 2) {
      outb[lc] = (__bf16)(v + addsrc[lc]);
    } else if constexpr (MODE == 3) {
      outb[lc] = (__bf16)fmaxf(v, 0.f);
    } else {
      outf[((size_t)b * C + c) * L + l] = v;
    }
  }
}

// ---------------------------------------------------------------------------
// Flash attention.  Q,Kt: (B, L, 512) bf16 (row l, contiguous channels).
// Vcm: (B, 512, L) bf16 (channel-major).  Oout: (B, L, 512) f32.
// One WG (8 waves) = (b, h, 128-query block); wave w owns query rows w*16..+16.
// Iterate 16 key blocks of 64; S strip via WMMA -> LDS; online softmax by
// 128 row-owner threads; O rescale; P(LDS) x V via WMMA.
// ---------------------------------------------------------------------------
__global__ void __launch_bounds__(256) attn_kernel(
    const __bf16* __restrict__ Q, const __bf16* __restrict__ Kt,
    const __bf16* __restrict__ Vcm, float* __restrict__ Oout) {
  constexpr int L = 1024, CH = 512, DH = 64;
  __shared__ float S[128][65];
  __shared__ __align__(16) __bf16 P[128][72];
  __shared__ float corr[128];
  __shared__ float linv[128];

  int tid = threadIdx.x;
  int wave = tid >> 5, lane = tid & 31, r = lane & 15, hi = lane >> 4;
  int qblk = blockIdx.x & 7;
  int h = (blockIdx.x >> 3) & 7;
  int b = blockIdx.x >> 6;
  int q0 = qblk * 128;
  int rowb = wave * 16 + hi * 8;  // first of the 8 query rows this lane holds

  const __bf16* qptr =
      Q + ((size_t)b * L + q0 + wave * 16 + r) * CH + h * DH + hi * 16;
  v16bf aq0 = load_a_frag(qptr);       // d = 0..31
  v16bf aq1 = load_a_frag(qptr + 32);  // d = 32..63

  v8f o0 = {}, o1 = {}, o2 = {}, o3 = {};
  float m_i = -3.0e38f, l_i = 0.0f;  // meaningful only for tid < 128

  for (int j = 0; j < 16; ++j) {
    int key0 = j * 64;
    // ---- S strip: 16 query rows x 64 keys ----
#pragma unroll
    for (int kt = 0; kt < 4; ++kt) {
      const __bf16* kp =
          Kt + ((size_t)b * L + key0 + kt * 16 + r) * CH + h * DH + hi * 16;
      v8f s = {};
      s = wmma_bf16(aq0, load_b_frag(kp), s);
      s = wmma_bf16(aq1, load_b_frag(kp + 32), s);
#pragma unroll
      for (int i = 0; i < 8; ++i)
        S[rowb + i][kt * 16 + r] = s[i] * 0.125f;  // 1/sqrt(64)
    }
    __syncthreads();
    // ---- online softmax: thread tid owns query row tid ----
    if (tid < 128) {
      float mx = m_i;
      for (int m = 0; m < 64; ++m) mx = fmaxf(mx, S[tid][m]);
      float cf = __expf(m_i - mx);
      float sum = 0.f;
      for (int m = 0; m < 64; ++m) {
        float p = __expf(S[tid][m] - mx);
        sum += p;
        P[tid][m] = (__bf16)p;
      }
      l_i = l_i * cf + sum;
      m_i = mx;
      corr[tid] = cf;
    }
    __syncthreads();
    // ---- rescale O rows ----
    {
      float f[8];
#pragma unroll
      for (int i = 0; i < 8; ++i) f[i] = corr[rowb + i];
#pragma unroll
      for (int i = 0; i < 8; ++i) {
        o0[i] *= f[i];
        o1[i] *= f[i];
        o2[i] *= f[i];
        o3[i] *= f[i];
      }
    }
    // ---- O += P x V^T : A = P (query x key), B = Vcm rows (d, contiguous m)
#pragma unroll
    for (int ks = 0; ks < 2; ++ks) {
      const __bf16* pp = &P[wave * 16 + r][ks * 32 + hi * 16];
      v16bf ap = load_a_frag(pp);
      const __bf16* vb =
          Vcm + ((size_t)b * CH + h * DH + r) * L + key0 + ks * 32 + hi * 16;
      o0 = wmma_bf16(ap, load_b_frag(vb), o0);
      o1 = wmma_bf16(ap, load_b_frag(vb + (size_t)16 * L), o1);
      o2 = wmma_bf16(ap, load_b_frag(vb + (size_t)32 * L), o2);
      o3 = wmma_bf16(ap, load_b_frag(vb + (size_t)48 * L), o3);
    }
  }
  if (tid < 128) linv[tid] = 1.0f / l_i;
  __syncthreads();
#pragma unroll
  for (int i = 0; i < 8; ++i) {
    float inv = linv[rowb + i];
    float* op = Oout + ((size_t)b * L + q0 + rowb + i) * CH + h * DH + r;
    op[0]  = o0[i] * inv;
    op[16] = o1[i] * inv;
    op[32] = o2[i] * inv;
    op[48] = o3[i] * inv;
  }
}

// ---------------------------------------------------------------------------
// Host-side orchestration
// ---------------------------------------------------------------------------
extern "C" void kernel_launch(void* const* d_in, const int* in_sizes, int n_in,
                              void* d_out, int out_size, void* d_ws,
                              size_t ws_size, hipStream_t stream) {
  (void)in_sizes; (void)n_in; (void)out_size; (void)ws_size;
  constexpr int B = 8, Cin = 256, Cout = 512, L = 1024, Cff = 1024;

  const float* x_query = (const float*)d_in[0];
  const float* x_kv    = (const float*)d_in[1];
  const float* wq      = (const float*)d_in[2];
  const float* wk      = (const float*)d_in[4];
  const float* wv      = (const float*)d_in[6];
  const float* w1      = (const float*)d_in[8];
  const float* w2      = (const float*)d_in[10];
  float* out = (float*)d_out;

  char* ws = (char*)d_ws;
  size_t off = 0;
  auto carve = [&](size_t bytes) -> void* {
    void* p = ws + off;
    off = (off + bytes + 255) & ~(size_t)255;
    return p;
  };
  __bf16* wq_b   = (__bf16*)carve((size_t)Cout * Cout * 2);
  __bf16* wk_b   = (__bf16*)carve((size_t)Cout * Cin * 2);
  __bf16* wv_b   = (__bf16*)carve((size_t)Cout * Cin * 2);
  __bf16* w1_b   = (__bf16*)carve((size_t)Cff * Cout * 2);
  __bf16* w2_b   = (__bf16*)carve((size_t)Cout * Cff * 2);
  float*  xq_t_f = (float*)carve((size_t)B * L * Cout * 4);
  __bf16* xq_t_b = (__bf16*)carve((size_t)B * L * Cout * 2);
  __bf16* xkv_t_b = (__bf16*)carve((size_t)B * L * Cin * 2);
  float*  gbuf   = (float*)carve((size_t)B * L * Cff * 4);  // reused gemm out
  __bf16* q_b    = (__bf16*)carve((size_t)B * L * Cout * 2);
  __bf16* k_b    = (__bf16*)carve((size_t)B * L * Cout * 2);
  __bf16* v_b    = (__bf16*)carve((size_t)B * Cout * L * 2);  // channel-major
  float*  projf  = (float*)carve((size_t)B * L * Cout * 4);
  __bf16* z_b    = (__bf16*)carve((size_t)B * L * Cout * 2);
  __bf16* h_b    = (__bf16*)carve((size_t)B * L * Cff * 2);

  // 1) weights -> bf16
  cvt_bf16_kernel<<<(Cout * Cout + 255) / 256, 256, 0, stream>>>(wq, wq_b, Cout * Cout);
  cvt_bf16_kernel<<<(Cout * Cin + 255) / 256, 256, 0, stream>>>(wk, wk_b, Cout * Cin);
  cvt_bf16_kernel<<<(Cout * Cin + 255) / 256, 256, 0, stream>>>(wv, wv_b, Cout * Cin);
  cvt_bf16_kernel<<<(Cff * Cout + 255) / 256, 256, 0, stream>>>(w1, w1_b, Cff * Cout);
  cvt_bf16_kernel<<<(Cout * Cff + 255) / 256, 256, 0, stream>>>(w2, w2_b, Cout * Cff);

  // 2) transpose activations to (L, C)
  dim3 tb(32, 8);
  transpose_cvt_kernel<<<dim3(L / 32, Cout / 32, B), tb, 0, stream>>>(
      x_query, xq_t_f, xq_t_b, Cout, L);
  transpose_cvt_kernel<<<dim3(L / 32, Cin / 32, B), tb, 0, stream>>>(
      x_kv, nullptr, xkv_t_b, Cin, L);

  auto gemm_blocks = [](int M, int N) { return (8 * (M / 16) * (N / 64)) / 8; };

  // 3) q = inorm(Wq x)     -> (B,L,512) bf16
  gemm_bf16_kernel<<<gemm_blocks(Cout, L), 256, 0, stream>>>(wq_b, xq_t_b, gbuf, Cout, L, Cout);
  inorm_kernel<0><<<dim3(Cout / 256, B), 256, 0, stream>>>(gbuf, nullptr, nullptr, q_b, Cout);
  // 4) k = inorm(Wk x_kv)  -> (B,L,512) bf16
  gemm_bf16_kernel<<<gemm_blocks(Cout, L), 256, 0, stream>>>(wk_b, xkv_t_b, gbuf, Cout, L, Cin);
  inorm_kernel<0><<<dim3(Cout / 256, B), 256, 0, stream>>>(gbuf, nullptr, nullptr, k_b, Cout);
  // 5) v = inorm(Wv x_kv)  -> (B,512,L) bf16 channel-major
  gemm_bf16_kernel<<<gemm_blocks(Cout, L), 256, 0, stream>>>(wv_b, xkv_t_b, gbuf, Cout, L, Cin);
  inorm_kernel<1><<<dim3(Cout / 256, B), 256, 0, stream>>>(gbuf, nullptr, nullptr, v_b, Cout);

  // 6) attention -> projf (B,L,512) f32
  attn_kernel<<<B * 8 * (L / 128), 256, 0, stream>>>(q_b, k_b, v_b, projf);

  // 7) z = inorm(proj) + x_query -> bf16 (B,L,512)
  inorm_kernel<2><<<dim3(Cout / 256, B), 256, 0, stream>>>(projf, xq_t_f, nullptr, z_b, Cout);

  // 8) h = relu(inorm(W1 z)) -> bf16 (B,L,1024)
  gemm_bf16_kernel<<<gemm_blocks(Cff, L), 256, 0, stream>>>(w1_b, z_b, gbuf, Cff, L, Cout);
  inorm_kernel<3><<<dim3(Cff / 256, B), 256, 0, stream>>>(gbuf, nullptr, nullptr, h_b, Cff);

  // 9) y = inorm(W2 h) -> f32 (B,512,L) into d_out
  gemm_bf16_kernel<<<gemm_blocks(Cout, L), 256, 0, stream>>>(w2_b, h_b, gbuf, Cout, L, Cff);
  inorm_kernel<4><<<dim3(Cout / 256, B), 256, 0, stream>>>(gbuf, nullptr, out, nullptr, Cout);
}
